// CLRerHead_70686571757749
// MI455X (gfx1250) — compile-verified, matrix operand order checked by reading, not run on hardware
//
#include <hip/hip_runtime.h>
#include <math.h>

#define NRR 72
#define NSS 36
#define NPP 192
#define BB  32
#define FCC 64
#define MROWS (BB*NPP)     /* 6144 */
#define KFC   (64*NSS)     /* 2304 */

typedef __attribute__((ext_vector_type(16))) _Float16 v16h;
typedef __attribute__((ext_vector_type(8)))  _Float16 v8h;
typedef __attribute__((ext_vector_type(8)))  float    v8f;

__device__ __forceinline__ int sidx(int j) { return (j * 71) / 35; }  // SAMPLE_IDX

// ---------------------------------------------------------------------------
// WMMA 16x64 tile with manual double-buffered K-pipeline (peeled epilogue,
// branch-free steady state so both operand sets stay live in distinct regs).
// A: (16xK f16 row-major), Bt: (NxK f16 row-major). K multiple of 64.
// CDNA5 16-bit A/B VGPR layout (CDNA5 ISA): lane L holds row (L&15);
// elements 0..7 -> K = off..off+7, 8..15 -> K = off+16..off+23, off=8*(L>>4).
// ---------------------------------------------------------------------------
struct KTile { v8h alo, ahi, bl[4], bh[4]; };

__device__ __forceinline__ void load_tile(const _Float16* __restrict__ Ap,
                                          const _Float16* __restrict__ Bp,
                                          int ldb, int k0, KTile& t)
{
  t.alo = *(const v8h*)(Ap + k0);
  t.ahi = *(const v8h*)(Ap + k0 + 16);
#pragma unroll
  for (int j = 0; j < 4; ++j) {
    const _Float16* bj = Bp + (size_t)j * 16 * ldb + k0;
    t.bl[j] = *(const v8h*)(bj);
    t.bh[j] = *(const v8h*)(bj + 16);
  }
}

__device__ __forceinline__ void compute_tile(const KTile& t, v8f acc[4])
{
  v16h a;
#pragma unroll
  for (int i = 0; i < 8; ++i) { a[i] = t.alo[i]; a[i + 8] = t.ahi[i]; }
#pragma unroll
  for (int j = 0; j < 4; ++j) {
    v16h b;
#pragma unroll
    for (int i = 0; i < 8; ++i) { b[i] = t.bl[j][i]; b[i + 8] = t.bh[j][i]; }
    acc[j] = __builtin_amdgcn_wmma_f32_16x16x32_f16(false, a, false, b,
                                                    (short)0, acc[j], false, false);
  }
}

__device__ __forceinline__ void wmma_tile4(const _Float16* __restrict__ A,
                                           const _Float16* __restrict__ Bt,
                                           int lda, int ldb, int K,
                                           int m0, int n0, v8f acc[4])
{
  const int lane = threadIdx.x & 31;
  const int hh   = lane >> 4;
  const int l16  = lane & 15;
  const int koff = hh * 8;
  const _Float16* Ap = A + (size_t)(m0 + l16) * lda + koff;
  const _Float16* Bp = Bt + (size_t)(n0 + l16) * ldb + koff;

  KTile t0, t1;
  load_tile(Ap, Bp, ldb, 0, t0);                 // prologue
  int k0 = 0;
  for (; k0 + 64 < K; k0 += 64) {                // branch-free steady state
    load_tile(Ap, Bp, ldb, k0 + 32, t1);
    compute_tile(t0, acc);
    load_tile(Ap, Bp, ldb, k0 + 64, t0);
    compute_tile(t1, acc);
  }
  load_tile(Ap, Bp, ldb, k0 + 32, t1);           // peeled epilogue
  compute_tile(t0, acc);
  compute_tile(t1, acc);
}

// ---------------------------------------------------------------------------
// Convert fp32 W (KxN) -> f16 transposed Wt (NxK)
// ---------------------------------------------------------------------------
__global__ void k_cvtT(const float* __restrict__ W, _Float16* __restrict__ Wt,
                       int K, int N)
{
  int idx = blockIdx.x * blockDim.x + threadIdx.x;
  if (idx >= K * N) return;
  int k = idx / N, n = idx % N;
  Wt[(size_t)n * K + k] = (_Float16)W[idx];
}

// ---------------------------------------------------------------------------
// Time embedding: sin_emb -> gelu MLP -> time_emb -> scale/shift + token
// One block (256 thr) per batch element.
// ---------------------------------------------------------------------------
__global__ void k_time(const int* __restrict__ t,
                       const float* __restrict__ Wt1, const float* __restrict__ bt1,
                       const float* __restrict__ Wt2, const float* __restrict__ bt2,
                       const float* __restrict__ Wst, const float* __restrict__ bst,
                       const float* __restrict__ Wtc, const float* __restrict__ btc,
                       float* __restrict__ scale, float* __restrict__ shift,
                       float* __restrict__ token)
{
  int b = blockIdx.x, tid = threadIdx.x;
  __shared__ float se[64], emb[256], te[256], ts[256];
  if (tid < 32) {
    float fr  = __expf(-(float)tid * (9.210340371976184f / 31.0f)); // log(1e4)/31
    float ang = (float)t[b] * fr;
    se[tid]      = __sinf(ang);
    se[tid + 32] = __cosf(ang);
  }
  __syncthreads();
  {
    float s = bt1[tid];
    for (int i = 0; i < 64; ++i) s += se[i] * Wt1[i * 256 + tid];
    float x = s;
    emb[tid] = 0.5f * x * (1.f + tanhf(0.7978845608028654f * (x + 0.044715f * x * x * x)));
  }
  __syncthreads();
  {
    float s = bt2[tid];
    for (int i = 0; i < 256; ++i) s += emb[i] * Wt2[i * 256 + tid];
    te[tid] = s;
    ts[tid] = s / (1.f + __expf(-s));   // silu
  }
  __syncthreads();
  if (tid < 128) {
    float s = bst[tid];
    for (int i = 0; i < 256; ++i) s += ts[i] * Wst[i * 128 + tid];
    if (tid < 64) scale[b * 64 + tid] = s;
    else          shift[b * 64 + tid - 64] = s;
  }
  if (tid < 64) {
    float s = btc[tid];
    for (int i = 0; i < 256; ++i) s += te[i] * Wtc[i * 64 + tid];
    token[b * 64 + tid] = s;
  }
}

// ---------------------------------------------------------------------------
// Anchor update + gen_xs + sampled_xs (+ final output assembly)
// reg==nullptr => init pass (copy anchors from inputs, compute samples).
// ---------------------------------------------------------------------------
__global__ void k_update(const float* __restrict__ anchor_in,
                         float* __restrict__ anchor_out,
                         const float* __restrict__ reg,
                         const float* __restrict__ cls,
                         float* __restrict__ sampled,
                         float* __restrict__ out, int last)
{
  int idx = blockIdx.x * blockDim.x + threadIdx.x;
  if (idx >= MROWS) return;
  float sy = anchor_in[idx * 3 + 0];
  float sx = anchor_in[idx * 3 + 1];
  float th = anchor_in[idx * 3 + 2];
  if (reg) {
    sy += reg[(size_t)idx * 76 + 0];
    sx += reg[(size_t)idx * 76 + 1];
    th += reg[(size_t)idx * 76 + 2];
  }
  anchor_out[idx * 3 + 0] = sy;
  anchor_out[idx * 3 + 1] = sx;
  anchor_out[idx * 3 + 2] = th;
  float tt    = tanf(th * 3.14159265358979323846f);
  float tsafe = (fabsf(tt) < 0.001f) ? 0.001f : tt;
  float coef  = 0.4f / tsafe;                 // (IMG_H/IMG_W)/tan
  float xs[NRR];
  for (int r = 0; r < NRR; ++r) xs[r] = sx + ((float)r / 71.0f - sy) * coef;
  for (int s = 0; s < NSS; ++s) sampled[(size_t)idx * NSS + s] = xs[sidx(35 - s)];
  if (last) {
    float* o = out + (size_t)idx * 78;
    o[0] = cls[idx * 2 + 0];
    o[1] = cls[idx * 2 + 1];
    o[2] = sy; o[3] = sx; o[4] = th;
    o[5] = reg[(size_t)idx * 76 + 3];
    for (int r = 0; r < NRR; ++r) o[6 + r] = xs[r] + reg[(size_t)idx * 76 + 4 + r];
  }
}

// ---------------------------------------------------------------------------
// Bilinear ROI pooling: feat(B,64,H,W) sampled at 36 points -> roi f16 (6144,2304)
// roi[bp][c*36+s]. One block per (b,p).
// ---------------------------------------------------------------------------
__global__ void k_pool(const float* __restrict__ feat, int H, int W,
                       const float* __restrict__ sampled,
                       _Float16* __restrict__ roi)
{
  int bp = blockIdx.x;
  int b  = bp / NPP;
  int tid = threadIdx.x;
  __shared__ int   sI[NSS][4];
  __shared__ float sWt[NSS][4];
  if (tid < NSS) {
    int s = tid;
    float fy = (float)sidx(35 - s) / 71.0f;             // PRIOR_FEAT_YS
    float y  = fy * (float)(H - 1);                     // (gy+1)/2*(H-1)
    float x  = sampled[(size_t)bp * NSS + s] * (float)(W - 1);
    float x0 = floorf(x), y0 = floorf(y);
    float wx1 = x - x0, wy1 = y - y0;
    float cx[2] = { x0, x0 + 1.f }, cy[2] = { y0, y0 + 1.f };
    float wx[2] = { 1.f - wx1, wx1 }, wy[2] = { 1.f - wy1, wy1 };
    for (int j = 0; j < 2; ++j)
      for (int i = 0; i < 2; ++i) {
        float fx = cx[i], fyy = cy[j];
        bool valid = (fx >= 0.f) && (fx <= (float)(W - 1)) &&
                     (fyy >= 0.f) && (fyy <= (float)(H - 1));
        int ix = (int)fminf(fmaxf(fx, 0.f), (float)(W - 1));
        int iy = (int)fminf(fmaxf(fyy, 0.f), (float)(H - 1));
        sI[s][j * 2 + i]  = iy * W + ix;
        sWt[s][j * 2 + i] = valid ? wx[i] * wy[j] : 0.f;
      }
  }
  __syncthreads();
  const size_t fb0 = (size_t)b * 64 * H * W;
  for (int e = tid; e < KFC; e += 256) {
    int c = e / NSS, s = e % NSS;
    const float* fb = feat + fb0 + (size_t)c * H * W;
    float v = sWt[s][0] * fb[sI[s][0]] + sWt[s][1] * fb[sI[s][1]]
            + sWt[s][2] * fb[sI[s][2]] + sWt[s][3] * fb[sI[s][3]];
    roi[(size_t)bp * KFC + e] = (_Float16)v;
  }
}

// ---------------------------------------------------------------------------
// f = relu(roi @ W_fc + b_fc) + time_token[b]   (WMMA, K=2304, 16x64/wave)
// ---------------------------------------------------------------------------
__global__ void __launch_bounds__(32, 1)
k_fc(const _Float16* __restrict__ roi, const _Float16* __restrict__ Wt,
     const float* __restrict__ bias, const float* __restrict__ token,
     float* __restrict__ f, _Float16* __restrict__ fh)
{
  int m0 = blockIdx.x * 16;
  v8f acc[4] = {};
  wmma_tile4(roi, Wt, KFC, KFC, KFC, m0, 0, acc);
  int lane = threadIdx.x & 31, hh = lane >> 4, nl = lane & 15;
#pragma unroll
  for (int j = 0; j < 4; ++j) {
    int n = j * 16 + nl;
#pragma unroll
    for (int r = 0; r < 8; ++r) {
      int m = m0 + r + 8 * hh;
      int b = m / NPP;
      float v = fmaxf(acc[j][r] + bias[n], 0.f) + token[b * 64 + n];
      f[(size_t)m * 64 + n]  = v;
      fh[(size_t)m * 64 + n] = (_Float16)v;
    }
  }
}

// ---------------------------------------------------------------------------
// Generic 64x64 projection (WMMA, K=64, 16x64/wave) with optional bias/relu
// and f32 / f16 / per-batch-transposed-f16 outputs.
// ---------------------------------------------------------------------------
__global__ void __launch_bounds__(32, 1)
k_gemm64(const _Float16* __restrict__ A, const _Float16* __restrict__ Bt,
         const float* __restrict__ bias, int relu,
         float* __restrict__ outF, _Float16* __restrict__ outH,
         _Float16* __restrict__ outT)
{
  int m0 = blockIdx.x * 16;
  v8f acc[4] = {};
  wmma_tile4(A, Bt, 64, 64, 64, m0, 0, acc);
  int lane = threadIdx.x & 31, hh = lane >> 4, nl = lane & 15;
#pragma unroll
  for (int j = 0; j < 4; ++j) {
    int n = j * 16 + nl;
#pragma unroll
    for (int r = 0; r < 8; ++r) {
      int m = m0 + r + 8 * hh;
      float v = acc[j][r];
      if (bias) v += bias[n];
      if (relu) v = fmaxf(v, 0.f);
      if (outF) outF[(size_t)m * 64 + n] = v;
      if (outH) outH[(size_t)m * 64 + n] = (_Float16)v;
      if (outT) {               // vT[b][n][p] for attention's B-operand
        int b = m / NPP, p = m % NPP;
        outT[(size_t)b * 64 * NPP + (size_t)n * NPP + p] = (_Float16)v;
      }
    }
  }
}

// ---------------------------------------------------------------------------
// scores[b,p,q] = (q_row . k_row)/8 : both operands row-major 192x64 per batch,
// so k serves directly as the NxK "Bt" operand. Each wave: 16 rows x 64 cols.
// ---------------------------------------------------------------------------
__global__ void __launch_bounds__(32, 1)
k_scores(const _Float16* __restrict__ q, const _Float16* __restrict__ k,
         float* __restrict__ scores)
{
  int b   = blockIdx.x;
  int m0  = b * NPP + blockIdx.y * 16;
  int nqb = blockIdx.z * 64;              // 0, 64, 128
  v8f acc[4] = {};
  wmma_tile4(q, k, 64, 64, 64, m0, b * NPP + nqb, acc);
  int lane = threadIdx.x & 31, hh = lane >> 4, nl = lane & 15;
#pragma unroll
  for (int j = 0; j < 4; ++j) {
    int nq = nqb + j * 16 + nl;
#pragma unroll
    for (int r = 0; r < 8; ++r) {
      int m = m0 + r + 8 * hh;
      scores[(size_t)m * NPP + nq] = acc[j][r] * 0.125f;   // 1/sqrt(64)
    }
  }
}

// ---------------------------------------------------------------------------
// Row softmax over 192 columns -> attn f16
// ---------------------------------------------------------------------------
__global__ void k_softmax(const float* __restrict__ scores, _Float16* __restrict__ attn)
{
  int row = blockIdx.x, tid = threadIdx.x;
  __shared__ float red[256];
  float v = (tid < NPP) ? scores[(size_t)row * NPP + tid] : -3.4e38f;
  red[tid] = v; __syncthreads();
  for (int s = 128; s > 0; s >>= 1) {
    if (tid < s) red[tid] = fmaxf(red[tid], red[tid + s]);
    __syncthreads();
  }
  float mx = red[0]; __syncthreads();
  float e = (tid < NPP) ? __expf(v - mx) : 0.f;
  red[tid] = e; __syncthreads();
  for (int s = 128; s > 0; s >>= 1) {
    if (tid < s) red[tid] += red[tid + s];
    __syncthreads();
  }
  float inv = 1.f / red[0];
  if (tid < NPP) attn[(size_t)row * NPP + tid] = (_Float16)(e * inv);
}

// ---------------------------------------------------------------------------
// av = attn(192x192) @ v(192x64) per batch  (Bt = vT, K=192, 16x64/wave)
// ---------------------------------------------------------------------------
__global__ void __launch_bounds__(32, 1)
k_av(const _Float16* __restrict__ attn, const _Float16* __restrict__ vT,
     _Float16* __restrict__ av)
{
  int b  = blockIdx.x;
  int m0 = b * NPP + blockIdx.y * 16;
  v8f acc[4] = {};
  wmma_tile4(attn, vT + (size_t)b * 64 * NPP, NPP, NPP, NPP, m0, 0, acc);
  int lane = threadIdx.x & 31, hh = lane >> 4, nl = lane & 15;
#pragma unroll
  for (int j = 0; j < 4; ++j) {
    int n = j * 16 + nl;
#pragma unroll
    for (int r = 0; r < 8; ++r) {
      int m = m0 + r + 8 * hh;
      av[(size_t)m * 64 + n] = (_Float16)acc[j][r];
    }
  }
}

// ---------------------------------------------------------------------------
// f2 = (f + av @ W_o) * (scale+1) + shift   (16x64/wave)
// ---------------------------------------------------------------------------
__global__ void __launch_bounds__(32, 1)
k_oproj(const _Float16* __restrict__ av, const _Float16* __restrict__ Wt,
        const float* __restrict__ f, const float* __restrict__ scale,
        const float* __restrict__ shift,
        float* __restrict__ f2, _Float16* __restrict__ f2h)
{
  int m0 = blockIdx.x * 16;
  v8f acc[4] = {};
  wmma_tile4(av, Wt, 64, 64, 64, m0, 0, acc);
  int lane = threadIdx.x & 31, hh = lane >> 4, nl = lane & 15;
#pragma unroll
  for (int j = 0; j < 4; ++j) {
    int n = j * 16 + nl;
#pragma unroll
    for (int r = 0; r < 8; ++r) {
      int m = m0 + r + 8 * hh;
      int b = m / NPP;
      float v = f[(size_t)m * 64 + n] + acc[j][r];
      v = v * (scale[b * 64 + n] + 1.f) + shift[b * 64 + n];
      f2[(size_t)m * 64 + n]  = v;
      f2h[(size_t)m * 64 + n] = (_Float16)v;
    }
  }
}

// ---------------------------------------------------------------------------
// Tiny fp32 heads (N=2 cls, N=76 reg) — scalar, full precision for geometry.
// ---------------------------------------------------------------------------
__global__ void k_head(const float* __restrict__ A, const float* __restrict__ W,
                       const float* __restrict__ bias, float* __restrict__ out, int N)
{
  int idx = blockIdx.x * blockDim.x + threadIdx.x;
  if (idx >= MROWS * N) return;
  int m = idx / N, n = idx % N;
  float s = bias[n];
  for (int kk = 0; kk < 64; ++kk) s += A[(size_t)m * 64 + kk] * W[kk * N + n];
  out[idx] = s;
}

// ---------------------------------------------------------------------------
extern "C" void kernel_launch(void* const* d_in, const int* in_sizes, int n_in,
                              void* d_out, int out_size, void* d_ws, size_t ws_size,
                              hipStream_t stream)
{
  (void)in_sizes; (void)n_in; (void)out_size; (void)ws_size;

  const float* feat0 = (const float*)d_in[0];
  const float* feat1 = (const float*)d_in[1];
  const float* feat2 = (const float*)d_in[2];
  const float* inputs = (const float*)d_in[3];
  const int*   t      = (const int*)d_in[4];
  const float* W_t1 = (const float*)d_in[5];
  const float* W_t2 = (const float*)d_in[6];
  const float* W_st = (const float*)d_in[7];
  const float* W_tc = (const float*)d_in[8];
  const float* W_fc = (const float*)d_in[9];
  const float* W64[8] = { (const float*)d_in[10], (const float*)d_in[11],
                          (const float*)d_in[12], (const float*)d_in[13],
                          (const float*)d_in[14], (const float*)d_in[15],
                          (const float*)d_in[16], (const float*)d_in[17] }; // q,k,v,o,c1,c2,r1,r2
  const float* W_cls = (const float*)d_in[18];
  const float* W_reg = (const float*)d_in[19];
  const float* b_t1 = (const float*)d_in[20];
  const float* b_t2 = (const float*)d_in[21];
  const float* b_st = (const float*)d_in[22];
  const float* b_tc = (const float*)d_in[23];
  const float* b_fc = (const float*)d_in[24];
  const float* b_c1 = (const float*)d_in[25];
  const float* b_c2 = (const float*)d_in[26];
  const float* b_r1 = (const float*)d_in[27];
  const float* b_r2 = (const float*)d_in[28];
  const float* b_cls = (const float*)d_in[29];
  const float* b_reg = (const float*)d_in[30];
  float* out = (float*)d_out;

  char* base = (char*)d_ws;
  size_t off = 0;
  auto alloc = [&](size_t bytes) -> void* {
    void* p = base + off;
    off = (off + bytes + 255) & ~(size_t)255;
    return p;
  };

  float*     scale   = (float*)alloc(BB * 64 * 4);
  float*     shiftv  = (float*)alloc(BB * 64 * 4);
  float*     token   = (float*)alloc(BB * 64 * 4);
  _Float16*  wt_fc   = (_Float16*)alloc((size_t)KFC * 64 * 2);
  _Float16*  wt64    = (_Float16*)alloc((size_t)8 * 64 * 64 * 2);
  float*     anchor  = (float*)alloc((size_t)MROWS * 3 * 4);
  float*     sampled = (float*)alloc((size_t)MROWS * NSS * 4);
  _Float16*  roi_h   = (_Float16*)alloc((size_t)MROWS * KFC * 2);
  float*     f       = (float*)alloc((size_t)MROWS * 64 * 4);
  _Float16*  f_h     = (_Float16*)alloc((size_t)MROWS * 64 * 2);
  _Float16*  q_h     = (_Float16*)alloc((size_t)MROWS * 64 * 2);
  _Float16*  k_h     = (_Float16*)alloc((size_t)MROWS * 64 * 2);
  _Float16*  vT_h    = (_Float16*)alloc((size_t)MROWS * 64 * 2);
  float*     scores  = (float*)alloc((size_t)MROWS * NPP * 4);
  _Float16*  attn_h  = (_Float16*)alloc((size_t)MROWS * NPP * 2);
  _Float16*  av_h    = (_Float16*)alloc((size_t)MROWS * 64 * 2);
  float*     f2      = (float*)alloc((size_t)MROWS * 64 * 4);
  _Float16*  f2_h    = (_Float16*)alloc((size_t)MROWS * 64 * 2);
  _Float16*  c1_h    = (_Float16*)alloc((size_t)MROWS * 64 * 2);
  float*     c2      = (float*)alloc((size_t)MROWS * 64 * 4);
  _Float16*  r1_h    = (_Float16*)alloc((size_t)MROWS * 64 * 2);
  float*     r2      = (float*)alloc((size_t)MROWS * 64 * 4);
  float*     cls     = (float*)alloc((size_t)MROWS * 2 * 4);
  float*     reg     = (float*)alloc((size_t)MROWS * 76 * 4);

  // --- one-time (per call) weight conversion to f16 transposed -------------
  k_cvtT<<<(KFC * 64 + 255) / 256, 256, 0, stream>>>(W_fc, wt_fc, KFC, 64);
  for (int i = 0; i < 8; ++i)
    k_cvtT<<<(64 * 64 + 255) / 256, 256, 0, stream>>>(W64[i], wt64 + (size_t)i * 64 * 64, 64, 64);
  _Float16* wt_q  = wt64 + 0 * 4096;
  _Float16* wt_k  = wt64 + 1 * 4096;
  _Float16* wt_v  = wt64 + 2 * 4096;
  _Float16* wt_o  = wt64 + 3 * 4096;
  _Float16* wt_c1 = wt64 + 4 * 4096;
  _Float16* wt_c2 = wt64 + 5 * 4096;
  _Float16* wt_r1 = wt64 + 6 * 4096;
  _Float16* wt_r2 = wt64 + 7 * 4096;

  // --- time embedding ------------------------------------------------------
  k_time<<<BB, 256, 0, stream>>>(t, W_t1, b_t1, W_t2, b_t2, W_st, b_st, W_tc, b_tc,
                                 scale, shiftv, token);

  // --- init anchors + sampled xs from inputs -------------------------------
  k_update<<<(MROWS + 255) / 256, 256, 0, stream>>>(inputs, anchor, nullptr, nullptr,
                                                    sampled, nullptr, 0);

  const float* feats[3] = { feat0, feat1, feat2 };
  const int Hs[3] = { 40, 20, 10 };
  const int Ws[3] = { 100, 50, 25 };

  const dim3 gFull(MROWS / 16);            // 384 waves, 16x64 tile each
  const dim3 gAttn(BB, NPP / 16, 3);       // 16 rows x 64 cols of 192
  const dim3 gAv(BB, NPP / 16);

  for (int it = 0; it < 3; ++it) {
    int lvl = 2 - it;                       // feat2 -> feat1 -> feat0

    k_pool<<<MROWS, 256, 0, stream>>>(feats[lvl], Hs[lvl], Ws[lvl], sampled, roi_h);

    k_fc<<<gFull, 32, 0, stream>>>(roi_h, wt_fc, b_fc, token, f, f_h);

    k_gemm64<<<gFull, 32, 0, stream>>>(f_h, wt_q, nullptr, 0, nullptr, q_h,  nullptr);
    k_gemm64<<<gFull, 32, 0, stream>>>(f_h, wt_k, nullptr, 0, nullptr, k_h,  nullptr);
    k_gemm64<<<gFull, 32, 0, stream>>>(f_h, wt_v, nullptr, 0, nullptr, nullptr, vT_h);

    k_scores<<<gAttn, 32, 0, stream>>>(q_h, k_h, scores);
    k_softmax<<<MROWS, 256, 0, stream>>>(scores, attn_h);
    k_av<<<gAv, 32, 0, stream>>>(attn_h, vT_h, av_h);
    k_oproj<<<gFull, 32, 0, stream>>>(av_h, wt_o, f, scale, shiftv, f2, f2_h);

    k_gemm64<<<gFull, 32, 0, stream>>>(f2_h, wt_c1, b_c1, 1, nullptr, c1_h, nullptr);
    k_gemm64<<<gFull, 32, 0, stream>>>(c1_h, wt_c2, b_c2, 1, c2, nullptr, nullptr);
    k_gemm64<<<gFull, 32, 0, stream>>>(f2_h, wt_r1, b_r1, 1, nullptr, r1_h, nullptr);
    k_gemm64<<<gFull, 32, 0, stream>>>(r1_h, wt_r2, b_r2, 1, r2, nullptr, nullptr);

    k_head<<<(MROWS * 2 + 255) / 256, 256, 0, stream>>>(c2, W_cls, b_cls, cls, 2);
    k_head<<<(MROWS * 76 + 255) / 256, 256, 0, stream>>>(r2, W_reg, b_reg, reg, 76);

    k_update<<<(MROWS + 255) / 256, 256, 0, stream>>>(anchor, anchor, reg, cls,
                                                      sampled, out, it == 2 ? 1 : 0);
  }
}
